// SAPP_39118562132321
// MI455X (gfx1250) — compile-verified, matrix operand order checked by reading, not run on hardware
//
#include <hip/hip_runtime.h>
#include <hip/hip_bf16.h>
#include <stdint.h>

#define BATCH 8
#define NPTS  8192
#define DFEAT 64
#define SPTS  (NPTS / 2)   // 4096
#define KNB   16
#define CIN   67
#define K0P   96           // CIN padded to multiple of 32
#define SMP   4            // samples per wave in MLP kernel
#define MWAVES 2           // waves per MLP block

typedef __attribute__((ext_vector_type(16))) __bf16 v16bf;
typedef __attribute__((ext_vector_type(8)))  float  v8f;

// 16-bit WMMA A/B per-lane element -> K index mapping (ISA 7.12.2):
// lanes 0-15: e0..7 -> K0..7,  e8..15 -> K16..23
// lanes16-31: e0..7 -> K8..15, e8..15 -> K24..31
__device__ __forceinline__ int kmap(int e, int hi) {
  return e + ((e >= 8) ? 8 : 0) + hi * 8;
}

// ---------------------------------------------------------------------------
// 1) Farthest point sampling: one block per batch, 1024 threads (32 waves),
//    dist[] resident in LDS. Serial over 4096 iterations -> minimize latency.
// ---------------------------------------------------------------------------
__global__ void __launch_bounds__(1024)
fps_kernel(const float* __restrict__ xyz, int* __restrict__ fpsIdx) {
  __shared__ float dist[NPTS];          // 32 KB
  __shared__ float redD[32];
  __shared__ int   redI[32];
  __shared__ float cent[3];
  __shared__ int   sFar;
  const int b   = blockIdx.x;
  const int tid = threadIdx.x;
  const float* X = xyz + (size_t)b * 3 * NPTS;

  for (int n = tid; n < NPTS; n += 1024) dist[n] = 1e10f;
  if (tid == 0) sFar = 0;
  __syncthreads();

  for (int it = 0; it < SPTS; ++it) {
    const int far = sFar;
    if (tid == 0) {
      fpsIdx[b * SPTS + it] = far;       // emit BEFORE update (matches scan)
      cent[0] = X[far]; cent[1] = X[NPTS + far]; cent[2] = X[2 * NPTS + far];
    }
    __syncthreads();
    const float cx = cent[0], cy = cent[1], cz = cent[2];
    float bd = -1.0f; int bi = 0;
    #pragma unroll
    for (int rep = 0; rep < NPTS / 1024; ++rep) {
      int n = tid + rep * 1024;
      float dx = X[n] - cx, dy = X[NPTS + n] - cy, dz = X[2 * NPTS + n] - cz;
      float d  = dx * dx + dy * dy + dz * dz;
      float nd = fminf(dist[n], d);
      dist[n]  = nd;
      if (nd > bd) { bd = nd; bi = n; }  // ascending n -> first-idx ties kept
    }
    #pragma unroll
    for (int off = 16; off > 0; off >>= 1) {
      float od = __shfl_down(bd, off);
      int   oi = __shfl_down(bi, off);
      if (od > bd || (od == bd && oi < bi)) { bd = od; bi = oi; }
    }
    if ((tid & 31) == 0) { redD[tid >> 5] = bd; redI[tid >> 5] = bi; }
    __syncthreads();
    if (tid < 32) {
      bd = redD[tid]; bi = redI[tid];
      #pragma unroll
      for (int off = 16; off > 0; off >>= 1) {
        float od = __shfl_down(bd, off);
        int   oi = __shfl_down(bi, off);
        if (od > bd || (od == bd && oi < bi)) { bd = od; bi = oi; }
      }
      if (tid == 0) sFar = bi;
    }
    __syncthreads();
  }
}

// ---------------------------------------------------------------------------
// 2) new_xyz output: out0[b][c][s] = xyz[b][c][fpsIdx[b][s]]
// ---------------------------------------------------------------------------
__global__ void newxyz_kernel(const float* __restrict__ xyz,
                              const int* __restrict__ fpsIdx,
                              float* __restrict__ out0) {
  int t = blockIdx.x * blockDim.x + threadIdx.x;
  if (t >= BATCH * SPTS) return;
  int b = t / SPTS, s = t % SPTS;
  const float* X = xyz + (size_t)b * 3 * NPTS;
  int fi = fpsIdx[b * SPTS + s];
  out0[((size_t)b * 3 + 0) * SPTS + s] = X[fi];
  out0[((size_t)b * 3 + 1) * SPTS + s] = X[NPTS + fi];
  out0[((size_t)b * 3 + 2) * SPTS + s] = X[2 * NPTS + fi];
}

// ---------------------------------------------------------------------------
// 3) kNN: one block (128 thr) per centroid; register top-16 + LDS selection.
// ---------------------------------------------------------------------------
__global__ void __launch_bounds__(128)
knn_kernel(const float* __restrict__ xyz, const int* __restrict__ fpsIdx,
           int* __restrict__ knnIdx) {
  __shared__ float cd[128 * KNB];       // 8 KB
  __shared__ int   ci[128 * KNB];       // 8 KB
  const int bs  = blockIdx.x;
  const int b   = bs / SPTS, s = bs % SPTS;
  const int tid = threadIdx.x;
  const float* X = xyz + (size_t)b * 3 * NPTS;
  const int   fi = fpsIdx[b * SPTS + s];
  const float cx = X[fi], cy = X[NPTS + fi], cz = X[2 * NPTS + fi];

  float bd[KNB]; int bi[KNB];
  #pragma unroll
  for (int j = 0; j < KNB; ++j) { bd[j] = 3.0e38f; bi[j] = 0x7fffffff; }

  for (int n = tid; n < NPTS; n += 128) {
    float dx = X[n] - cx, dy = X[NPTS + n] - cy, dz = X[2 * NPTS + n] - cz;
    float d  = dx * dx + dy * dy + dz * dz;
    if (d < bd[KNB - 1]) {               // fully unrolled insertion (no scratch)
      bd[KNB - 1] = d; bi[KNB - 1] = n;
      #pragma unroll
      for (int j = KNB - 1; j > 0; --j) {
        if (bd[j] < bd[j - 1]) {
          float td = bd[j]; bd[j] = bd[j - 1]; bd[j - 1] = td;
          int   ti = bi[j]; bi[j] = bi[j - 1]; bi[j - 1] = ti;
        }
      }
    }
  }
  #pragma unroll
  for (int j = 0; j < KNB; ++j) { cd[tid * KNB + j] = bd[j]; ci[tid * KNB + j] = bi[j]; }
  __syncthreads();

  if (tid < 32) {                        // wave 0: 16 argmin selection passes
    for (int sel = 0; sel < KNB; ++sel) {
      float md = 3.0e38f; int mi = 0x7fffffff; int mp = 0;
      for (int p = tid; p < 128 * KNB; p += 32) {
        float d = cd[p]; int i = ci[p];
        if (d < md || (d == md && i < mi)) { md = d; mi = i; mp = p; }
      }
      #pragma unroll
      for (int off = 16; off > 0; off >>= 1) {
        float od = __shfl_down(md, off);
        int   oi = __shfl_down(mi, off);
        int   op = __shfl_down(mp, off);
        if (od < md || (od == md && oi < mi)) { md = od; mi = oi; mp = op; }
      }
      mi = __shfl(mi, 0);
      mp = __shfl(mp, 0);
      if (tid == 0) knnIdx[(size_t)bs * KNB + sel] = mi;
      cd[mp] = 3.0e38f;                  // all lanes write same value/addr
      __builtin_amdgcn_wave_barrier();
    }
  }
}

// ---------------------------------------------------------------------------
// 4) Weight prep: f32 [Cout,Cin] -> bf16 B-fragment layout, K padded w/ zeros.
//    Frag block (kt,nt) = 512 bf16: [lane 0..31][elem 0..15], elem->K via kmap.
// ---------------------------------------------------------------------------
__global__ void prep_kernel(const float* __restrict__ W0, const float* __restrict__ W1,
                            const float* __restrict__ W2, const float* __restrict__ W3,
                            __bf16* __restrict__ wf) {
  int id = blockIdx.x * blockDim.x + threadIdx.x;
  if (id >= 51200) return;
  const float* W; int Cin, NT, base;
  if (id < 6144)       { W = W0; Cin = CIN; NT = 4;  base = 0;     }  // 96x64
  else if (id < 10240) { W = W1; Cin = 64;  NT = 4;  base = 6144;  }  // 64x64
  else if (id < 18432) { W = W2; Cin = 64;  NT = 8;  base = 10240; }  // 64x128
  else                 { W = W3; Cin = 128; NT = 16; base = 18432; }  // 128x256
  int r = id - base;
  int blk = r / 512, rem = r % 512;
  int lane = rem / 16, e = rem % 16;
  int kt = blk / NT, nt = blk % NT;
  int k = kt * 32 + kmap(e, lane >> 4);
  int n = nt * 16 + (lane & 15);
  float v = (k < Cin) ? W[n * Cin + k] : 0.0f;
  wf[id] = (__bf16)v;
}

// ---------------------------------------------------------------------------
// 5) Feature transpose: f32 [B,D,N] -> bf16 [B,N,D] (row-gather friendly).
//    LDS-tiled: 32 points x 64 channels per block.
// ---------------------------------------------------------------------------
__global__ void __launch_bounds__(256)
transpose_kernel(const float* __restrict__ feat, __bf16* __restrict__ featT) {
  __shared__ float tile[DFEAT][33];      // +1 pad vs bank conflicts
  const int blk = blockIdx.x;
  const int b  = blk / (NPTS / 32);
  const int n0 = (blk % (NPTS / 32)) * 32;
  const float* F = feat + (size_t)b * DFEAT * NPTS;
  const int tid = threadIdx.x;
  for (int t = tid; t < DFEAT * 32; t += 256) {   // coalesced over n
    int c = t >> 5, n = t & 31;
    tile[c][n] = F[(size_t)c * NPTS + n0 + n];
  }
  __syncthreads();
  // 32 rows x 64 bf16; each thread emits one 16B chunk (8 bf16)
  {
    int n = tid >> 3, ch = tid & 7;
    union { __bf16 h[8]; uint4 u; } pk;
    #pragma unroll
    for (int j = 0; j < 8; ++j) pk.h[j] = (__bf16)tile[ch * 8 + j][n];
    *(uint4*)(featT + ((size_t)b * NPTS + n0 + n) * DFEAT + ch * 8) = pk.u;
  }
}

// ---------------------------------------------------------------------------
// 6) Fused 4-layer MLP + K-maxpool. One wave per 4 samples; B-frag reuse x4.
// ---------------------------------------------------------------------------
template<int KD, int ND, int SSRC, int SDST>
__device__ __forceinline__ void mlp_layer4(const __bf16* src, __bf16* dst,
                                           const __bf16* __restrict__ wf,
                                           const float* __restrict__ bias, int lane) {
  const int KT = KD / 32, NT = ND / 16;
  const int m = lane & 15, hi = lane >> 4;
  #pragma unroll
  for (int nt = 0; nt < NT; ++nt) {
    v8f acc[SMP];
    #pragma unroll
    for (int p = 0; p < SMP; ++p) acc[p] = (v8f){0.f,0.f,0.f,0.f,0.f,0.f,0.f,0.f};
    #pragma unroll
    for (int kt = 0; kt < KT; ++kt) {
      union { uint4 u[2]; v16bf v; } ub;           // load B fragment ONCE
      const __bf16* bp = wf + (size_t)(kt * NT + nt) * 512 + lane * 16;
      ub.u[0] = ((const uint4*)bp)[0];
      ub.u[1] = ((const uint4*)bp)[1];
      #pragma unroll
      for (int p = 0; p < SMP; ++p) {              // reuse across 4 samples
        union { uint4 u[2]; v16bf v; } ua;
        const __bf16* ap = src + p * SSRC + m * KD + kt * 32 + hi * 8;
        ua.u[0] = *(const uint4*)(ap);
        ua.u[1] = *(const uint4*)(ap + 16);
        acc[p] = __builtin_amdgcn_wmma_f32_16x16x32_bf16(false, ua.v, false, ub.v,
                                                         (short)0, acc[p], false, false);
      }
    }
    float bv = bias[nt * 16 + m];
    #pragma unroll
    for (int p = 0; p < SMP; ++p) {
      #pragma unroll
      for (int r = 0; r < 8; ++r) {
        float v = acc[p][r] + bv;
        v = v > 0.0f ? v : 0.0f;
        dst[p * SDST + (r + hi * 8) * ND + nt * 16 + m] = (__bf16)v;
      }
    }
  }
}

template<int KD, int ND, int SSRC>
__device__ __forceinline__ void mlp_final4(const __bf16* src,
                                           const __bf16* __restrict__ wf,
                                           const float* __restrict__ bias, int lane,
                                           float* __restrict__ out1, int b,
                                           const int* s) {
  const int KT = KD / 32, NT = ND / 16;
  const int m = lane & 15, hi = lane >> 4;
  #pragma unroll
  for (int nt = 0; nt < NT; ++nt) {
    v8f acc[SMP];
    #pragma unroll
    for (int p = 0; p < SMP; ++p) acc[p] = (v8f){0.f,0.f,0.f,0.f,0.f,0.f,0.f,0.f};
    #pragma unroll
    for (int kt = 0; kt < KT; ++kt) {
      union { uint4 u[2]; v16bf v; } ub;
      const __bf16* bp = wf + (size_t)(kt * NT + nt) * 512 + lane * 16;
      ub.u[0] = ((const uint4*)bp)[0];
      ub.u[1] = ((const uint4*)bp)[1];
      #pragma unroll
      for (int p = 0; p < SMP; ++p) {
        union { uint4 u[2]; v16bf v; } ua;
        const __bf16* ap = src + p * SSRC + m * KD + kt * 32 + hi * 8;
        ua.u[0] = *(const uint4*)(ap);
        ua.u[1] = *(const uint4*)(ap + 16);
        acc[p] = __builtin_amdgcn_wmma_f32_16x16x32_bf16(false, ua.v, false, ub.v,
                                                         (short)0, acc[p], false, false);
      }
    }
    float bv = bias[nt * 16 + m];
    #pragma unroll
    for (int p = 0; p < SMP; ++p) {
      float mx = 0.0f;                     // relu outputs >= 0
      #pragma unroll
      for (int r = 0; r < 8; ++r) {
        float v = acc[p][r] + bv;
        v = v > 0.0f ? v : 0.0f;
        mx = fmaxf(mx, v);                 // max over rows M=r+hi*8 (neighbors)
      }
      float om = __shfl_xor(mx, 16);       // combine M 0-7 with M 8-15
      mx = fmaxf(mx, om);
      if (hi == 0)
        out1[((size_t)b * 256 + nt * 16 + m) * SPTS + s[p]] = mx;
    }
  }
}

__global__ void __launch_bounds__(MWAVES * 32)
mlp_kernel(const float* __restrict__ xyz, const __bf16* __restrict__ featT,
           const float* __restrict__ b0, const float* __restrict__ b1,
           const float* __restrict__ b2, const float* __restrict__ b3,
           const int* __restrict__ fpsIdx, const int* __restrict__ knnIdx,
           const __bf16* __restrict__ wf, float* __restrict__ out1) {
  __shared__ __align__(16) __bf16 sb0[MWAVES][SMP][16 * K0P];   // 24 KB
  __shared__ __align__(16) __bf16 sb1[MWAVES][SMP][16 * 128];   // 32 KB
  __shared__ int snid[MWAVES][SMP][KNB];
  const int tid  = threadIdx.x;
  const int w    = tid >> 5, lane = tid & 31;
  const int sgBase = blockIdx.x * (MWAVES * SMP) + w * SMP;     // 4096%8==0 -> same b
  const int b = sgBase / SPTS;
  int s[SMP];
  #pragma unroll
  for (int p = 0; p < SMP; ++p) s[p] = (sgBase + p) % SPTS;

  const float*  X  = xyz + (size_t)b * 3 * NPTS;
  const __bf16* FT = featT + (size_t)b * NPTS * DFEAT;

  #pragma unroll
  for (int p = 0; p < SMP; ++p)
    if (lane < KNB) snid[w][p][lane] = knnIdx[(size_t)(sgBase + p) * KNB + lane];
  __syncthreads();

  // Build X0 tiles: [16 rows x 96 ch] = [feat(64) | dxyz(3) | 0 pad]
  #pragma unroll
  for (int p = 0; p < SMP; ++p) {
    const int fi = fpsIdx[b * SPTS + s[p]];
    const float cx = X[fi], cy = X[NPTS + fi], cz = X[2 * NPTS + fi];
    // 64 feat channels per row: 8 x b128 vector copies
    for (int t = lane; t < 16 * 8; t += 32) {
      int r = t >> 3, ch = t & 7;
      int n = snid[w][p][r];
      *(uint4*)(sb0[w][p] + r * K0P + ch * 8) =
          *((const uint4*)(FT + (size_t)n * DFEAT) + ch);
    }
    // cols 64..95: xyz diff + zero pad (c = 64 + lane, r sweeps 0..15)
    for (int t = lane; t < 16 * 32; t += 32) {
      int r = t >> 5, c = 64 + (t & 31);
      int n = snid[w][p][r];
      float v = 0.0f;
      if (c == 64)      v = X[n] - cx;
      else if (c == 65) v = X[NPTS + n] - cy;
      else if (c == 66) v = X[2 * NPTS + n] - cz;
      sb0[w][p][r * K0P + c] = (__bf16)v;
    }
  }
  __syncthreads();
  mlp_layer4<K0P, 64, 16 * K0P, 16 * 128>(sb0[w][0], sb1[w][0], wf + 0,     b0, lane);
  __syncthreads();
  mlp_layer4<64, 64, 16 * 128, 16 * K0P>(sb1[w][0], sb0[w][0], wf + 6144,  b1, lane);
  __syncthreads();
  mlp_layer4<64, 128, 16 * K0P, 16 * 128>(sb0[w][0], sb1[w][0], wf + 10240, b2, lane);
  __syncthreads();
  mlp_final4<128, 256, 16 * 128>(sb1[w][0], wf + 18432, b3, lane, out1, b, s);
}

// ---------------------------------------------------------------------------
extern "C" void kernel_launch(void* const* d_in, const int* in_sizes, int n_in,
                              void* d_out, int out_size, void* d_ws, size_t ws_size,
                              hipStream_t stream) {
  (void)in_sizes; (void)n_in; (void)out_size; (void)ws_size;
  const float* xyz  = (const float*)d_in[0];
  const float* feat = (const float*)d_in[1];
  const float* W0 = (const float*)d_in[2]; const float* b0 = (const float*)d_in[3];
  const float* W1 = (const float*)d_in[4]; const float* b1 = (const float*)d_in[5];
  const float* W2 = (const float*)d_in[6]; const float* b2 = (const float*)d_in[7];
  const float* W3 = (const float*)d_in[8]; const float* b3 = (const float*)d_in[9];
  float* out = (float*)d_out;

  char* ws = (char*)d_ws;
  size_t off = 0;
  int*    fpsIdx = (int*)(ws + off);    off += (size_t)BATCH * SPTS * 4;            // 128 KB
  int*    knnIdx = (int*)(ws + off);    off += (size_t)BATCH * SPTS * KNB * 4;      // 2 MB
  __bf16* wf     = (__bf16*)(ws + off); off += 51200 * 2; off = (off + 15) & ~15ull;// 100 KB
  __bf16* featT  = (__bf16*)(ws + off);                                             // 8 MB

  float* out0 = out;                             // [B,3,S]
  float* out1 = out + (size_t)BATCH * 3 * SPTS;  // [B,256,S]

  prep_kernel<<<(51200 + 255) / 256, 256, 0, stream>>>(W0, W1, W2, W3, wf);
  transpose_kernel<<<BATCH * (NPTS / 32), 256, 0, stream>>>(feat, featT);
  fps_kernel<<<BATCH, 1024, 0, stream>>>(xyz, fpsIdx);
  newxyz_kernel<<<(BATCH * SPTS + 255) / 256, 256, 0, stream>>>(xyz, fpsIdx, out0);
  knn_kernel<<<BATCH * SPTS, 128, 0, stream>>>(xyz, fpsIdx, knnIdx);
  mlp_kernel<<<BATCH * SPTS / (MWAVES * SMP), MWAVES * 32, 0, stream>>>(
      xyz, featT, b0, b1, b2, b3, fpsIdx, knnIdx, wf, out1);
}
